// SeparableCritic_37538014167585
// MI455X (gfx1250) — matrix operand order repaired and checked by reading, not compiled, not run on hardware
//
#include <hip/hip_runtime.h>
#include <hip/hip_bf16.h>

// ---------------------------------------------------------------------------
// SeparableCritic on MI455X (gfx1250, wave32, WMMA)
//
// Roofline: output write = 8192*8192*4B = 256MB @ 23.3 TB/s ~= 11us floor.
// Total math ~29 GFLOP -> must run on v_wmma_f32_16x16x32_f16 (f16 in, f32
// accumulate) so the whole pipeline stays HBM-write bound. All activations
// kept f16 (L2-resident), final z vectors are L2-normalized so f16 operands
// with f32 accumulation give ~1e-3 absolute accuracy on the cosine outputs.
// ---------------------------------------------------------------------------

typedef __attribute__((ext_vector_type(16))) _Float16 v16h;
typedef __attribute__((ext_vector_type(8)))  _Float16 v8h;
typedef __attribute__((ext_vector_type(8)))  float    v8f;
typedef __attribute__((ext_vector_type(4)))  _Float16 half4;

#define N_PTS 8192
#define D_IN  64
#define HID   512
#define LAT   128
#define M_TOT 16384   // x and y stacked

union V16Pack { struct { v8h lo, hi; } p; v16h v; };
__device__ __forceinline__ v16h make_v16(v8h lo, v8h hi) {
  V16Pack u; u.p.lo = lo; u.p.hi = hi; return u.v;
}

// --------------------------------------------------------------------------
// Kernel 0: concat [x;y] f32 -> f16
// --------------------------------------------------------------------------
__global__ void concat_f16_kernel(const float* __restrict__ x,
                                  const float* __restrict__ y,
                                  _Float16* __restrict__ out) {
  int idx = blockIdx.x * blockDim.x + threadIdx.x;
  const int half = N_PTS * D_IN;
  float v = (idx < half) ? x[idx] : y[idx - half];
  out[idx] = (_Float16)v;
}

// --------------------------------------------------------------------------
// Generic WMMA GEMM: C[M x N] = act(A[M x K](f16) * B + bias)
//   B_T_F16=false : B is f32 [K x N] (weight matrix, transposed into LDS)
//   B_T_F16=true  : B is f16 [N x K] (zy rows == B columns; straight copy)
// Block: 256 threads = 8 waves; C-tile 128x128 (wave w owns tile-row w,
// 8 tiles of 16x16, A-fragment reused 8x). K chunked by 64 through LDS.
//
// ISA fragment layouts (CDNA5 ISA):
//   A 16x32 f16 : lane (m=l&15, kh=l>>4): elems 0..7 = K[8kh..8kh+7],
//                 elems 8..15 = K[16+8kh..16+8kh+7]  -> two 16B LDS reads
//   B 32x16 f16 : lane (n=l&15, kh=l>>4): elems 0..7 = K[8kh..8kh+7],
//                 elems 8..15 = K[16+8kh..16+8kh+7] of column n -> two 16B LDS reads
//   C 16x16 f32 : lane (n=l&15, mh=l>>4): vgpr r -> C[8*mh+r][n]
// --------------------------------------------------------------------------
template<int K, int N, bool B_T_F16, bool RELU, bool OUT_F16, bool HAS_BIAS>
__global__ __launch_bounds__(256) void gemm_wmma_kernel(
    const _Float16* __restrict__ A, const void* __restrict__ Bv,
    const float* __restrict__ bias, void* __restrict__ Outv) {
  constexpr int KC  = 64;
  constexpr int PAD = 8;   // row stride 72 f16 = 144B -> bank-friendly
  __shared__ alignas(16) _Float16 As[128][KC + PAD];
  __shared__ alignas(16) _Float16 Bs[128][KC + PAD];

  const int tid  = threadIdx.x;
  const int w    = tid >> 5;        // wave id 0..7 = C tile-row
  const int lane = tid & 31;
  const int l15  = lane & 15;       // m for A, n for B/C
  const int kh   = lane >> 4;       // K-half for A/B, M-half for C
  const int rowBase = blockIdx.y * 128;
  const int colBase = blockIdx.x * 128;

  v8f acc[8];
  const v8f vzero = {0.f, 0.f, 0.f, 0.f, 0.f, 0.f, 0.f, 0.f};
#pragma unroll
  for (int t = 0; t < 8; ++t) acc[t] = vzero;

  for (int kc = 0; kc < K; kc += KC) {
    // ---- cooperative tile loads ----
    for (int i = tid; i < 128 * KC; i += 256) {        // A: k fastest
      int r = i >> 6, c = i & 63;
      As[r][c] = A[(size_t)(rowBase + r) * K + kc + c];
    }
    if (B_T_F16) {
      const _Float16* Bh = (const _Float16*)Bv;
      for (int i = tid; i < 128 * KC; i += 256) {      // straight row copy
        int r = i >> 6, c = i & 63;
        Bs[r][c] = Bh[(size_t)(colBase + r) * K + kc + c];
      }
    } else {
      const float* Bf = (const float*)Bv;
      for (int i = tid; i < 128 * KC; i += 256) {      // n fastest: coalesced
        int n = i & 127, k = i >> 7;
        Bs[n][k] = (_Float16)Bf[(size_t)(kc + k) * N + colBase + n];
      }
    }
    __syncthreads();

    // ---- WMMA over this K-chunk ----
#pragma unroll
    for (int ks = 0; ks < KC; ks += 32) {
      v8h alo = *(const v8h*)&As[w * 16 + l15][ks + 8 * kh];
      v8h ahi = *(const v8h*)&As[w * 16 + l15][ks + 16 + 8 * kh];
      v16h af = make_v16(alo, ahi);
#pragma unroll
      for (int tn = 0; tn < 8; ++tn) {
        v8h blo = *(const v8h*)&Bs[tn * 16 + l15][ks + 8 * kh];
        v8h bhi = *(const v8h*)&Bs[tn * 16 + l15][ks + 16 + 8 * kh];
        v16h bf = make_v16(blo, bhi);
        acc[tn] = __builtin_amdgcn_wmma_f32_16x16x32_f16(
            false, af, false, bf, (short)0, acc[tn], false, false);
      }
    }
    __syncthreads();
  }

  // ---- epilogue: bias + relu + store ----
#pragma unroll
  for (int tn = 0; tn < 8; ++tn) {
    const int col = colBase + tn * 16 + l15;
    const float bv = HAS_BIAS ? bias[col] : 0.f;
#pragma unroll
    for (int r = 0; r < 8; ++r) {
      const int row = rowBase + w * 16 + kh * 8 + r;
      float v = acc[tn][r] + bv;
      if (RELU) v = fmaxf(v, 0.f);
      if (OUT_F16) {
        ((_Float16*)Outv)[(size_t)row * N + col] = (_Float16)v;
      } else {
        // final 256MB result: stream past caches
        __builtin_nontemporal_store(v, (float*)Outv + (size_t)row * N + col);
      }
    }
  }
}

// --------------------------------------------------------------------------
// Kernel: L2 normalize each 128-wide row (one wave32 per row), f32 -> f16
// --------------------------------------------------------------------------
__global__ void l2norm_kernel(const float* __restrict__ Zf,
                              _Float16* __restrict__ Zh) {
  const int gtid = blockIdx.x * blockDim.x + threadIdx.x;
  const int row  = gtid >> 5;
  const int lane = gtid & 31;
  const float4 v = ((const float4*)(Zf + (size_t)row * LAT))[lane];
  float s = v.x * v.x + v.y * v.y + v.z * v.z + v.w * v.w;
#pragma unroll
  for (int m = 16; m > 0; m >>= 1) s += __shfl_xor(s, m, 32);
  const float inv = 1.f / fmaxf(sqrtf(s), 1e-12f);
  half4 o = { (_Float16)(v.x * inv), (_Float16)(v.y * inv),
              (_Float16)(v.z * inv), (_Float16)(v.w * inv) };
  ((half4*)(Zh + (size_t)row * LAT))[lane] = o;
}

// --------------------------------------------------------------------------
// Launch
// --------------------------------------------------------------------------
extern "C" void kernel_launch(void* const* d_in, const int* in_sizes, int n_in,
                              void* d_out, int out_size, void* d_ws, size_t ws_size,
                              hipStream_t stream) {
  const float* x  = (const float*)d_in[0];
  const float* y  = (const float*)d_in[1];
  const float* W1 = (const float*)d_in[2];
  const float* b1 = (const float*)d_in[3];
  const float* W2 = (const float*)d_in[4];
  const float* b2 = (const float*)d_in[5];
  const float* W3 = (const float*)d_in[6];
  const float* b3 = (const float*)d_in[7];
  float* out = (float*)d_out;

  // workspace layout (46 MB total)
  char* ws = (char*)d_ws;
  _Float16* Xh = (_Float16*)(ws);                        //  2 MB [16384 x  64] f16
  _Float16* H1 = (_Float16*)(ws + ((size_t)2  << 20));   // 16 MB [16384 x 512] f16
  _Float16* H2 = (_Float16*)(ws + ((size_t)18 << 20));   // 16 MB [16384 x 512] f16
  float*    Zf = (float*)   (ws + ((size_t)34 << 20));   //  8 MB [16384 x 128] f32
  _Float16* Zh = (_Float16*)(ws + ((size_t)42 << 20));   //  4 MB [16384 x 128] f16

  // 0) concat+convert inputs
  concat_f16_kernel<<<dim3((M_TOT * D_IN) / 256), 256, 0, stream>>>(x, y, Xh);

  // 1) H1 = relu(X @ W1 + b1)      [16384x64]@[64x512]
  gemm_wmma_kernel<64, 512, false, true, true, true>
      <<<dim3(512 / 128, M_TOT / 128), 256, 0, stream>>>(Xh, W1, b1, H1);

  // 2) H2 = relu(H1 @ W2 + b2)     [16384x512]@[512x512]
  gemm_wmma_kernel<512, 512, false, true, true, true>
      <<<dim3(512 / 128, M_TOT / 128), 256, 0, stream>>>(H1, W2, b2, H2);

  // 3) Z = H2 @ W3 + b3            [16384x512]@[512x128], f32 out
  gemm_wmma_kernel<512, 128, false, false, false, true>
      <<<dim3(128 / 128, M_TOT / 128), 256, 0, stream>>>(H2, W3, b3, Zf);

  // 4) z = Z / max(||Z||, eps)  -> f16
  l2norm_kernel<<<dim3((M_TOT * 32) / 256), 256, 0, stream>>>(Zf, Zh);

  // 5) out = zx @ zy^T  [8192x128]@[128x8192], f32 NT stores (write-BW bound)
  gemm_wmma_kernel<128, 8192, true, false, false, false>
      <<<dim3(8192 / 128, N_PTS / 128), 256, 0, stream>>>(
          Zh, (const void*)(Zh + (size_t)N_PTS * LAT), nullptr, out);
}
